// SparseAttention_28046136442889
// MI455X (gfx1250) — matrix operand order repaired and checked by reading, not compiled, not run on hardware
//
#include <hip/hip_runtime.h>
#include <hip/hip_bf16.h>
#include <math.h>

// ---------------- problem constants ----------------
#define SQ   1024          // sequence length S
#define DIMX 512           // model dim
#define NH   8             // heads
#define DH   64            // head dim
#define CBX  32            // compress block
#define NB   (SQ/CBX)      // 32 blocks
#define SBX  32            // select block
#define NSEL 4
#define NMEM 4
#define WINX 64
#define CD   (CBX*DH)      // 2048
#define NCK  (NMEM+NB)     // 36 compressed keys
#define SCL  0.125f        // DH^-0.5
#define EPSR 1.1920929e-07f
#define FNEG (-3.402823466e38f)

typedef __attribute__((ext_vector_type(16))) __bf16 v16bf;
typedef __attribute__((ext_vector_type(8)))  float  v8f;

// =====================================================================
// WMMA bf16 GEMM:  C[M,N] = A[M,K] (row major) * BT[N,K]^T (+bias)(+relu)
// One wave per 16x64 C tile; A fragment reused across 4 WMMAs.
// Software-pipelined: register double-buffered fragments, unrolled x2
// (requires K % 64 == 0) so next step's 10 b128 loads overlap the
// current step's 4 WMMAs (waits become s_wait_loadcnt <n>, not 0).
// Fragment layouts used here (wave32; A/B in contiguous k, not the ISA table order):
//   A 16x32 bf16: lane L -> row M=L%16, elem e -> K = e + 16*(L>=16)
//   B 32x16 bf16: lane L -> col N=L%16, elem e -> K = e + 16*(L>=16)
//   C/D f32:      vgpr r -> M = r + 8*(L>=16), N = L%16
// =====================================================================
__global__ __launch_bounds__(32)
void wmma_gemm_bf16(const __bf16* __restrict__ A, const __bf16* __restrict__ BT,
                    const float* __restrict__ bias,
                    float* __restrict__ Cf, __bf16* __restrict__ Cb,
                    int M, int N, int K, int relu)
{
    const int lane = threadIdx.x;
    const int hs   = lane >> 4;     // half-select (0: lanes 0-15, 1: lanes 16-31)
    const int lm   = lane & 15;
    const int n0   = blockIdx.x * 64;
    const int m0   = blockIdx.y * 16;

    const __bf16* aBase = A + (size_t)(m0 + lm) * K + hs * 16;
    const __bf16* bBase[4];
    #pragma unroll
    for (int t = 0; t < 4; ++t)
        bBase[t] = BT + (size_t)(n0 + t * 16 + lm) * K + hs * 16;

    auto loadA = [&](int k0, v16bf& a) {
        const __bf16* ap = aBase + k0;
        #pragma unroll
        for (int i = 0; i < 8; ++i) a[i] = ap[i];
        ap += 8;
        #pragma unroll
        for (int i = 0; i < 8; ++i) a[8 + i] = ap[i];
    };
    auto loadB = [&](int k0, v16bf* b) {
        #pragma unroll
        for (int t = 0; t < 4; ++t) {
            const __bf16* bp = bBase[t] + k0;
            #pragma unroll
            for (int i = 0; i < 16; ++i) b[t][i] = bp[i];
        }
    };

    v8f zero = {};
    v8f acc[4];
    #pragma unroll
    for (int t = 0; t < 4; ++t) acc[t] = zero;

    v16bf a0, a1, b0[4], b1[4];
    loadA(0, a0); loadB(0, b0);

    for (int k0 = 0; k0 < K; k0 += 64) {
        // prefetch one stage ahead of the double buffer
        if (k0 + 128 < K) {
            __builtin_prefetch(aBase + k0 + 128, 0, 1);
            #pragma unroll
            for (int t = 0; t < 4; ++t) __builtin_prefetch(bBase[t] + k0 + 128, 0, 1);
        }
        // stage 1: issue loads for k0+32, compute k0
        loadA(k0 + 32, a1); loadB(k0 + 32, b1);
        #pragma unroll
        for (int t = 0; t < 4; ++t)
            acc[t] = __builtin_amdgcn_wmma_f32_16x16x32_bf16(
                false, a0, false, b0[t], (short)0, acc[t], false, false);
        // stage 2: issue loads for k0+64, compute k0+32
        if (k0 + 64 < K) { loadA(k0 + 64, a0); loadB(k0 + 64, b0); }
        #pragma unroll
        for (int t = 0; t < 4; ++t)
            acc[t] = __builtin_amdgcn_wmma_f32_16x16x32_bf16(
                false, a1, false, b1[t], (short)0, acc[t], false, false);
    }

    #pragma unroll
    for (int t = 0; t < 4; ++t) {
        int col = n0 + t * 16 + lm;
        float bv = bias ? bias[col] : 0.0f;
        #pragma unroll
        for (int r = 0; r < 8; ++r) {
            int row = m0 + r + hs * 8;
            float v = acc[t][r] + bv;
            if (relu) v = fmaxf(v, 0.0f);
            size_t o = (size_t)row * N + col;
            if (Cf) Cf[o] = v;
            if (Cb) Cb[o] = (__bf16)v;
        }
    }
}

// ---------------- weight convert + transpose: in[K,N] f32 -> out[N,K] bf16 ----
__global__ void transpose_to_bf16(const float* __restrict__ in, __bf16* __restrict__ out,
                                  int K, int N)
{
    int i = blockIdx.x * blockDim.x + threadIdx.x;
    if (i >= K * N) return;
    int k = i / N, n = i % N;
    out[(size_t)n * K + k] = (__bf16)in[i];
}

// ---------------- RMSNorm: x = inp * rsqrt(mean(inp^2)+eps) * g ----------------
__global__ __launch_bounds__(256)
void rmsnorm_kernel(const float* __restrict__ inp, const float* __restrict__ g,
                    float* __restrict__ xf, __bf16* __restrict__ xb)
{
    __shared__ float red[256];
    int s = blockIdx.x, t = threadIdx.x;
    const float* row = inp + (size_t)s * DIMX;
    float ss = 0.f;
    for (int i = t; i < DIMX; i += 256) { float v = row[i]; ss += v * v; }
    red[t] = ss; __syncthreads();
    for (int off = 128; off > 0; off >>= 1) {
        if (t < off) red[t] += red[t + off];
        __syncthreads();
    }
    float scale = rsqrtf(red[0] / DIMX + EPSR);
    for (int i = t; i < DIMX; i += 256) {
        float v = row[i] * scale * g[i];
        xf[(size_t)s * DIMX + i] = v;
        xb[(size_t)s * DIMX + i] = (__bf16)v;
    }
}

// ---------------- split qkv, build compress-MLP inputs (k/v blocks + pos) ------
__global__ void split_qkv_kernel(const float* __restrict__ qkv,
                                 const float* __restrict__ k_pos, const float* __restrict__ v_pos,
                                 float* __restrict__ q, float* __restrict__ k, float* __restrict__ v,
                                 __bf16* __restrict__ kb, __bf16* __restrict__ vb)
{
    int i = blockIdx.x * blockDim.x + threadIdx.x;   // NH*SQ*DH
    if (i >= NH * SQ * DH) return;
    int d = i % DH, s = (i / DH) % SQ, h = i / (DH * SQ);
    const float* rp = qkv + (size_t)s * (3 * NH * DH);
    float qv = rp[h * DH + d];
    float kv = rp[NH * DH + h * DH + d];
    float vv = rp[2 * NH * DH + h * DH + d];
    size_t o = ((size_t)h * SQ + s) * DH + d;
    q[o] = qv; k[o] = kv; v[o] = vv;
    int b = s / CBX, c = s % CBX;
    size_t ro = (size_t)(h * NB + b) * CD + c * DH + d;
    kb[ro] = (__bf16)(kv + k_pos[(h * CBX + c) * DH + d]);
    vb[ro] = (__bf16)(vv + v_pos[(h * CBX + c) * DH + d]);
}

// ---------------- RoPE on q and k ----------------
__global__ void rope_kernel(const float* __restrict__ q, const float* __restrict__ k,
                            float* __restrict__ rq, float* __restrict__ rk)
{
    int i = blockIdx.x * blockDim.x + threadIdx.x;   // NH*SQ*(DH/2)
    if (i >= NH * SQ * (DH / 2)) return;
    int p = i % (DH / 2), s = (i / (DH / 2)) % SQ, h = i / ((DH / 2) * SQ);
    float inv = expf(-(2.0f * p / DH) * logf(10000.0f));
    float ang = s * inv;
    float c = cosf(ang), sn = sinf(ang);
    size_t o = ((size_t)h * SQ + s) * DH + 2 * p;
    float a0 = q[o], a1 = q[o + 1];
    rq[o]     = a0 * c - a1 * sn;
    rq[o + 1] = a1 * c + a0 * sn;
    a0 = k[o]; a1 = k[o + 1];
    rk[o]     = a0 * c - a1 * sn;
    rk[o + 1] = a1 * c + a0 * sn;
}

// ---------------- compressed attention + top-k block selection -----------------
__global__ __launch_bounds__(64)
void cattn_kernel(const float* __restrict__ q, const float* __restrict__ ck,
                  const float* __restrict__ cv, const float* __restrict__ mem_kv,
                  float* __restrict__ c_out, int* __restrict__ sel_idx, int* __restrict__ sel_ok)
{
    __shared__ float kS[NCK * DH];
    __shared__ float vS[NCK * DH];
    int h = blockIdx.x, tile = blockIdx.y, t = threadIdx.x;
    for (int i = t; i < NCK * DH; i += 64) {
        int j = i / DH, d = i % DH;
        float kk, vv;
        if (j < NMEM) {
            kk = mem_kv[((0 * NH + h) * NMEM + j) * DH + d];
            vv = mem_kv[((1 * NH + h) * NMEM + j) * DH + d];
        } else {
            int b = j - NMEM;
            kk = ck[(size_t)(h * NB + b) * DH + d];
            vv = cv[(size_t)(h * NB + b) * DH + d];
        }
        kS[i] = kk; vS[i] = vv;
    }
    __syncthreads();

    int s = tile * 64 + t;
    const float* qp = q + ((size_t)h * SQ + s) * DH;
    float qv[DH];
    #pragma unroll
    for (int d = 0; d < DH; ++d) qv[d] = qp[d];

    float lg[NCK];
    float mx = FNEG;
    #pragma unroll
    for (int j = 0; j < NCK; ++j) {
        bool valid = (j < NMEM) || ((j - NMEM + 1) * CBX - 1 < s);
        float x = FNEG;
        if (valid) {
            float acc = 0.f;
            #pragma unroll
            for (int d = 0; d < DH; ++d) acc += qv[d] * kS[j * DH + d];
            x = acc * SCL;
        }
        lg[j] = x;
        if (x > mx) mx = x;
    }
    float sum = 0.f;
    #pragma unroll
    for (int j = 0; j < NCK; ++j) { float p = expf(lg[j] - mx); lg[j] = p; sum += p; }
    float inv = 1.f / sum;

    float acc[DH];
    #pragma unroll
    for (int d = 0; d < DH; ++d) acc[d] = 0.f;
    #pragma unroll
    for (int j = 0; j < NCK; ++j) {
        float p = lg[j] * inv;
        #pragma unroll
        for (int d = 0; d < DH; ++d) acc[d] += p * vS[j * DH + d];
    }
    for (int d = 0; d < DH; ++d) c_out[((size_t)h * SQ + s) * DH + d] = acc[d];

    // top-NSEL over block probabilities (first-max tie-break == jax top_k)
    float imp[NB];
    #pragma unroll
    for (int b = 0; b < NB; ++b) imp[b] = lg[NMEM + b] * inv;
    for (int n = 0; n < NSEL; ++n) {
        int bi = 0; float bv = imp[0];
        #pragma unroll
        for (int b = 1; b < NB; ++b) { if (imp[b] > bv) { bv = imp[b]; bi = b; } }
        size_t so = ((size_t)h * SQ + s) * NSEL + n;
        sel_idx[so] = bi;
        sel_ok[so]  = (bv > 1e-10f) ? 1 : 0;
        imp[bi] = -1.f;
    }
}

// ---------------- fine (gathered-block) attention, online softmax --------------
__global__ __launch_bounds__(128)
void fattn_kernel(const float* __restrict__ rq, const float* __restrict__ rk,
                  const float* __restrict__ v, const int* __restrict__ sel_idx,
                  const int* __restrict__ sel_ok, float* __restrict__ f_out)
{
    int i = blockIdx.x * blockDim.x + threadIdx.x;   // NH*SQ
    if (i >= NH * SQ) return;
    int s = i % SQ, h = i / SQ;
    const float* qp = rq + ((size_t)h * SQ + s) * DH;
    float qv[DH];
    #pragma unroll
    for (int d = 0; d < DH; ++d) qv[d] = qp[d] * SCL;

    float m = FNEG, l = 0.f, acc[DH];
    #pragma unroll
    for (int d = 0; d < DH; ++d) acc[d] = 0.f;

    for (int bi = 0; bi < NSEL + 1; ++bi) {
        int blk; int ok;
        if (bi < NSEL) {
            size_t so = (size_t)i * NSEL + bi;
            blk = sel_idx[so]; ok = sel_ok[so];
            if (!ok) continue;
        } else {
            blk = s / SBX;           // own block, causal-masked
        }
        int cmax = (bi == NSEL) ? (s % SBX) : (SBX - 1);
        for (int c = 0; c <= cmax; ++c) {
            const float* kp = rk + ((size_t)h * SQ + blk * SBX + c) * DH;
            const float* vp = v  + ((size_t)h * SQ + blk * SBX + c) * DH;
            float x = 0.f;
            #pragma unroll
            for (int d = 0; d < DH; ++d) x += qv[d] * kp[d];
            float nm = fmaxf(m, x);
            float corr = expf(m - nm);
            float p = expf(x - nm);
            l = l * corr + p;
            #pragma unroll
            for (int d = 0; d < DH; ++d) acc[d] = acc[d] * corr + p * vp[d];
            m = nm;
        }
    }
    float inv = 1.f / l;
    for (int d = 0; d < DH; ++d) f_out[((size_t)h * SQ + s) * DH + d] = acc[d] * inv;
}

// ---------------- sliding-window attention (j in [s-WIN, s]) -------------------
__global__ __launch_bounds__(128)
void swin_kernel(const float* __restrict__ rq, const float* __restrict__ rk,
                 const float* __restrict__ v, float* __restrict__ s_out)
{
    int i = blockIdx.x * blockDim.x + threadIdx.x;   // NH*SQ
    if (i >= NH * SQ) return;
    int s = i % SQ, h = i / SQ;
    const float* qp = rq + ((size_t)h * SQ + s) * DH;
    float qv[DH];
    #pragma unroll
    for (int d = 0; d < DH; ++d) qv[d] = qp[d] * SCL;

    float m = FNEG, l = 0.f, acc[DH];
    #pragma unroll
    for (int d = 0; d < DH; ++d) acc[d] = 0.f;

    int j0 = s - WINX; if (j0 < 0) j0 = 0;
    for (int j = j0; j <= s; ++j) {
        const float* kp = rk + ((size_t)h * SQ + j) * DH;
        const float* vp = v  + ((size_t)h * SQ + j) * DH;
        float x = 0.f;
        #pragma unroll
        for (int d = 0; d < DH; ++d) x += qv[d] * kp[d];
        float nm = fmaxf(m, x);
        float corr = expf(m - nm);
        float p = expf(x - nm);
        l = l * corr + p;
        #pragma unroll
        for (int d = 0; d < DH; ++d) acc[d] = acc[d] * corr + p * vp[d];
        m = nm;
    }
    float inv = 1.f / l;
    for (int d = 0; d < DH; ++d) s_out[((size_t)h * SQ + s) * DH + d] = acc[d] * inv;
}

// ---------------- strategy gates + combine -> bf16 for final GEMM --------------
__global__ __launch_bounds__(512)
void combine_kernel(const float* __restrict__ xf, const float* __restrict__ strat_w,
                    const float* __restrict__ strat_b,
                    const float* __restrict__ c_out, const float* __restrict__ f_out,
                    const float* __restrict__ s_out, __bf16* __restrict__ comb)
{
    __shared__ float st[3 * NH];
    int s = blockIdx.x, t = threadIdx.x;
    if (t < 3 * NH) {
        float a = strat_b[t];
        const float* xp = xf + (size_t)s * DIMX;
        for (int k2 = 0; k2 < DIMX; ++k2) a += xp[k2] * strat_w[k2 * (3 * NH) + t];
        st[t] = 1.f / (1.f + expf(-a));
    }
    __syncthreads();
    int h = t / DH, d = t % DH;
    size_t o = ((size_t)h * SQ + s) * DH + d;
    float val = st[h * 3 + 0] * c_out[o] + st[h * 3 + 1] * f_out[o] + st[h * 3 + 2] * s_out[o];
    comb[(size_t)s * DIMX + t] = (__bf16)val;
}

// =====================================================================
extern "C" void kernel_launch(void* const* d_in, const int* in_sizes, int n_in,
                              void* d_out, int out_size, void* d_ws, size_t ws_size,
                              hipStream_t stream)
{
    const float* inp     = (const float*)d_in[0];
    const float* norm_g  = (const float*)d_in[1];
    const float* w_qkv   = (const float*)d_in[2];
    const float* k_pos   = (const float*)d_in[3];
    const float* v_pos   = (const float*)d_in[4];
    const float* kc_w1   = (const float*)d_in[5];
    const float* kc_b1   = (const float*)d_in[6];
    const float* kc_w2   = (const float*)d_in[7];
    const float* kc_b2   = (const float*)d_in[8];
    const float* vc_w1   = (const float*)d_in[9];
    const float* vc_b1   = (const float*)d_in[10];
    const float* vc_w2   = (const float*)d_in[11];
    const float* vc_b2   = (const float*)d_in[12];
    const float* mem_kv  = (const float*)d_in[13];
    const float* strat_w = (const float*)d_in[14];
    const float* strat_b = (const float*)d_in[15];
    const float* out_w   = (const float*)d_in[16];

    // bump allocator over d_ws
    char* p = (char*)d_ws;
    auto alloc = [&](size_t bytes) -> char* {
        char* r = p; p += (bytes + 255) & ~(size_t)255; return r;
    };
    __bf16* wqkvT = (__bf16*)alloc((size_t)1536 * 512 * 2);
    __bf16* kcw1T = (__bf16*)alloc((size_t)2048 * 2048 * 2);
    __bf16* vcw1T = (__bf16*)alloc((size_t)2048 * 2048 * 2);
    __bf16* kcw2T = (__bf16*)alloc((size_t)64 * 2048 * 2);
    __bf16* vcw2T = (__bf16*)alloc((size_t)64 * 2048 * 2);
    __bf16* outwT = (__bf16*)alloc((size_t)512 * 512 * 2);
    float*  xf    = (float*) alloc((size_t)SQ * DIMX * 4);
    __bf16* xb    = (__bf16*)alloc((size_t)SQ * DIMX * 2);
    float*  qkv   = (float*) alloc((size_t)SQ * 3 * NH * DH * 4);
    float*  q     = (float*) alloc((size_t)NH * SQ * DH * 4);
    float*  k     = (float*) alloc((size_t)NH * SQ * DH * 4);
    float*  v     = (float*) alloc((size_t)NH * SQ * DH * 4);
    float*  rq    = (float*) alloc((size_t)NH * SQ * DH * 4);
    float*  rk    = (float*) alloc((size_t)NH * SQ * DH * 4);
    __bf16* kb    = (__bf16*)alloc((size_t)NH * NB * CD * 2);
    __bf16* vb    = (__bf16*)alloc((size_t)NH * NB * CD * 2);
    __bf16* hb    = (__bf16*)alloc((size_t)NH * NB * CD * 2);   // hidden (reused K then V)
    float*  ck    = (float*) alloc((size_t)NH * NB * DH * 4);
    float*  cv    = (float*) alloc((size_t)NH * NB * DH * 4);
    float*  c_o   = (float*) alloc((size_t)NH * SQ * DH * 4);
    float*  f_o   = (float*) alloc((size_t)NH * SQ * DH * 4);
    float*  s_o   = (float*) alloc((size_t)NH * SQ * DH * 4);
    int*    sidx  = (int*)   alloc((size_t)NH * SQ * NSEL * 4);
    int*    sok   = (int*)   alloc((size_t)NH * SQ * NSEL * 4);
    __bf16* comb  = (__bf16*)alloc((size_t)SQ * DIMX * 2);
    (void)ws_size; (void)in_sizes; (void)n_in; (void)out_size;

    const int T = 256;
    // weight conversion / transposition (bf16, [N,K] layout for WMMA B)
    transpose_to_bf16<<<(512 * 1536 + T - 1) / T, T, 0, stream>>>(w_qkv, wqkvT, 512, 1536);
    transpose_to_bf16<<<(2048 * 2048 + T - 1) / T, T, 0, stream>>>(kc_w1, kcw1T, 2048, 2048);
    transpose_to_bf16<<<(2048 * 2048 + T - 1) / T, T, 0, stream>>>(vc_w1, vcw1T, 2048, 2048);
    transpose_to_bf16<<<(2048 * 64 + T - 1) / T, T, 0, stream>>>(kc_w2, kcw2T, 2048, 64);
    transpose_to_bf16<<<(2048 * 64 + T - 1) / T, T, 0, stream>>>(vc_w2, vcw2T, 2048, 64);
    transpose_to_bf16<<<(512 * 512 + T - 1) / T, T, 0, stream>>>(out_w, outwT, 512, 512);

    rmsnorm_kernel<<<SQ, 256, 0, stream>>>(inp, norm_g, xf, xb);

    // QKV projection: [1024,512] x [512,1536]
    wmma_gemm_bf16<<<dim3(1536 / 64, SQ / 16), 32, 0, stream>>>(
        xb, wqkvT, nullptr, qkv, nullptr, SQ, 1536, 512, 0);

    split_qkv_kernel<<<(NH * SQ * DH + T - 1) / T, T, 0, stream>>>(
        qkv, k_pos, v_pos, q, k, v, kb, vb);

    // K-compress MLP: [256,2048]x[2048,2048] relu, then [256,2048]x[2048,64]
    wmma_gemm_bf16<<<dim3(2048 / 64, (NH * NB) / 16), 32, 0, stream>>>(
        kb, kcw1T, kc_b1, nullptr, hb, NH * NB, 2048, 2048, 1);
    wmma_gemm_bf16<<<dim3(64 / 64, (NH * NB) / 16), 32, 0, stream>>>(
        hb, kcw2T, kc_b2, ck, nullptr, NH * NB, 64, 2048, 0);
    // V-compress MLP (reuses hidden buffer)
    wmma_gemm_bf16<<<dim3(2048 / 64, (NH * NB) / 16), 32, 0, stream>>>(
        vb, vcw1T, vc_b1, nullptr, hb, NH * NB, 2048, 2048, 1);
    wmma_gemm_bf16<<<dim3(64 / 64, (NH * NB) / 16), 32, 0, stream>>>(
        hb, vcw2T, vc_b2, cv, nullptr, NH * NB, 64, 2048, 0);

    rope_kernel<<<(NH * SQ * (DH / 2) + T - 1) / T, T, 0, stream>>>(q, k, rq, rk);

    cattn_kernel<<<dim3(NH, SQ / 64), 64, 0, stream>>>(q, ck, cv, mem_kv, c_o, sidx, sok);
    fattn_kernel<<<(NH * SQ + 127) / 128, 128, 0, stream>>>(rq, rk, v, sidx, sok, f_o);
    swin_kernel<<<(NH * SQ + 127) / 128, 128, 0, stream>>>(rq, rk, v, s_o);

    combine_kernel<<<SQ, 512, 0, stream>>>(xf, strat_w, strat_b, c_o, f_o, s_o, comb);

    // output projection: [1024,512]x[512,512] -> d_out (f32)
    wmma_gemm_bf16<<<dim3(512 / 64, SQ / 16), 32, 0, stream>>>(
        comb, outwT, nullptr, (float*)d_out, nullptr, SQ, 512, 512, 0);
}
